// LISA_36017595744690
// MI455X (gfx1250) — matrix operand order repaired and checked by reading, not compiled, not run on hardware
//
#include <hip/hip_runtime.h>
#include <hip/hip_bf16.h>
#include <math.h>

typedef __attribute__((ext_vector_type(16))) _Float16 v16h;
typedef __attribute__((ext_vector_type(8)))  _Float16 v8h;
typedef __attribute__((ext_vector_type(8)))  float    v8f;

#define N_OBS  2048
#define D_DIM  128
#define RDIM   16
#define R_SAMP 500
#define R_PAD  512          // padded sample count (16 WMMA K-chunks of 32)
#define ZTS    520          // transposed f16 row stride in halves (bank padding)
#define SIGMA_C 0.01f
#define REG_C   0.8f

// ---- workspace layout (float offsets) ----
#define WS_G    0                       // G = H^T H           [16x16]
#define WS_W    256                     // W (gain core)       [16x16]
#define WS_M    512                     // m = alpha/alpha0    [16]
#define WS_T0   528                     // H^T EY              [16]
#define WS_TRC  544                     // relu(tr(LMMSE_C))+1e-6
#define WS_CP   545                     // lgamma(a0) - sum lgamma(a_k)
#define WS_DIRI 1024                    // diri                [N,16]
#define WS_HY   (WS_DIRI + N_OBS*RDIM)  // H^T y_n             [N,16]
#define WS_YY   (WS_HY   + N_OBS*RDIM)  // ||y_n||^2           [N]
#define WS_CQ   (WS_YY   + N_OBS)       // per-n proposal log-norm const [N]

// ---------------- RNG helpers ----------------
__device__ __forceinline__ unsigned long long mix64(unsigned long long x) {
    x += 0x9E3779B97F4A7C15ULL;
    x = (x ^ (x >> 30)) * 0xBF58476D1CE4E5B9ULL;
    x = (x ^ (x >> 27)) * 0x94D049BB133111EBULL;
    return x ^ (x >> 31);
}
__device__ __forceinline__ float u01(unsigned int u) {
    return ((float)(u >> 8) + 0.5f) * (1.0f / 16777216.0f);
}
// Marsaglia-Tsang gamma sampler (bounded attempts; boost for alpha<1)
__device__ __forceinline__ float gamma_sample(float alpha, unsigned long long& s) {
    float amod = (alpha < 1.0f) ? alpha + 1.0f : alpha;
    float d = amod - (1.0f / 3.0f);
    float c = rsqrtf(9.0f * d);
    float g = d;                          // fallback if all attempts rejected
    for (int it = 0; it < 8; ++it) {
        s = mix64(s);
        float u1 = u01((unsigned int)s);
        float u2 = u01((unsigned int)(s >> 32));
        float r  = sqrtf(fmaxf(-2.0f * logf(u1), 0.0f));
        float x  = r * cosf(6.28318530718f * u2);
        float v  = 1.0f + c * x;
        if (v <= 0.0f) continue;
        v = v * v * v;
        s = mix64(s);
        float u = u01((unsigned int)s);
        float xx = x * x;
        if (u < 1.0f - 0.0331f * xx * xx ||
            logf(u) < 0.5f * xx + d * (1.0f - v + logf(v))) {
            g = d * v;
            break;
        }
    }
    if (alpha < 1.0f) {
        s = mix64(s);
        g *= powf(u01((unsigned int)s), 1.0f / alpha);
    }
    return fmaxf(g, 1e-20f);
}

// ---------------- Kernel A: tiny-algebra setup (Woodbury) ----------------
__global__ void LISA_setup_kernel(const float* __restrict__ H,
                                  const float* __restrict__ alpha,
                                  float* __restrict__ ws) {
    __shared__ float m[RDIM];
    __shared__ float Cs[RDIM][RDIM], Gs[RDIM][RDIM], GC[RDIM][RDIM];
    __shared__ float T1[RDIM][RDIM], Wm[RDIM][RDIM];
    __shared__ float aug[RDIM][2 * RDIM];
    __shared__ float EY[D_DIM];
    __shared__ float a0s;
    const int t = threadIdx.x;
    const int k = t >> 4, l = t & 15;

    if (t == 0) {
        float a0 = 0.0f;
        for (int j = 0; j < RDIM; ++j) a0 += alpha[j];
        a0s = a0;
    }
    __syncthreads();
    if (t < RDIM) m[t] = alpha[t] / a0s;
    __syncthreads();

    // G = H^T H ; C = (diag(m) - m m^T)/(a0+1)
    {
        float s = 0.0f;
        for (int d = 0; d < D_DIM; ++d) s += H[d * RDIM + k] * H[d * RDIM + l];
        Gs[k][l] = s;
        Cs[k][l] = (((k == l) ? m[k] : 0.0f) - m[k] * m[l]) / (a0s + 1.0f);
    }
    __syncthreads();

    // GC = G*C ; Mmat = sigma*I + GC  (augmented with I for Gauss-Jordan)
    {
        float s = 0.0f;
        for (int j = 0; j < RDIM; ++j) s += Gs[k][j] * Cs[j][l];
        GC[k][l] = s;
        aug[k][l]        = ((k == l) ? SIGMA_C : 0.0f) + s;
        aug[k][l + RDIM] = (k == l) ? 1.0f : 0.0f;
    }
    __syncthreads();

    // 16x16 Gauss-Jordan (serial: trivially cheap)
    if (t == 0) {
        for (int p = 0; p < RDIM; ++p) {
            float inv = 1.0f / aug[p][p];
            for (int j = 0; j < 2 * RDIM; ++j) aug[p][j] *= inv;
            for (int r = 0; r < RDIM; ++r) {
                if (r == p) continue;
                float f = aug[r][p];
                for (int j = 0; j < 2 * RDIM; ++j) aug[r][j] -= f * aug[p][j];
            }
        }
    }
    __syncthreads();

    // T1 = GC * Minv
    {
        float s = 0.0f;
        for (int j = 0; j < RDIM; ++j) s += GC[k][j] * aug[j][RDIM + l];
        T1[k][l] = s;
    }
    __syncthreads();

    // W = (C - C*T1)/sigma   (gain = W * H^T, applied in 16-dim space)
    {
        float s = 0.0f;
        for (int j = 0; j < RDIM; ++j) s += Cs[k][j] * T1[j][l];
        float w = (Cs[k][l] - s) / SIGMA_C;
        Wm[k][l] = w;
        ws[WS_G + t] = Gs[k][l];
        ws[WS_W + t] = w;
    }
    __syncthreads();

    if (t == 0) {
        // trace(C - W*GC), relu + eps ;  global log-prior normalizer
        float tr = 0.0f;
        for (int kk = 0; kk < RDIM; ++kk) {
            float s = 0.0f;
            for (int j = 0; j < RDIM; ++j) s += Wm[kk][j] * GC[j][kk];
            tr += Cs[kk][kk] - s;
        }
        ws[WS_TRC] = fmaxf(tr, 0.0f) + 1e-6f;
        float lg = 0.0f;
        for (int j = 0; j < RDIM; ++j) lg += lgammaf(alpha[j]);
        ws[WS_CP] = lgammaf(a0s) - lg;
    }
    // EY = H m ; t0 = H^T EY
    if (t < D_DIM) {
        float s = 0.0f;
        for (int j = 0; j < RDIM; ++j) s += H[t * RDIM + j] * m[j];
        EY[t] = s;
    }
    __syncthreads();
    if (t < RDIM) {
        float s = 0.0f;
        for (int d = 0; d < D_DIM; ++d) s += H[d * RDIM + t] * EY[d];
        ws[WS_T0 + t] = s;
        ws[WS_M + t]  = m[t];
    }
}

// ---------------- Kernel B: per-observation LMMSE -> proposal params ----------------
__global__ void LISA_prep_kernel(const float* __restrict__ y,
                                 const float* __restrict__ H,
                                 float* __restrict__ ws) {
    const int n = blockIdx.x;
    const int t = threadIdx.x;
    __shared__ float hy[RDIM], lm[RDIM];
    __shared__ float yy;
    const float* yn = y + (size_t)n * D_DIM;

    if (t < RDIM) {
        float s = 0.0f;
        for (int d = 0; d < D_DIM; ++d) s += H[d * RDIM + t] * yn[d];
        hy[t] = s;
    } else if (t == RDIM) {
        float s = 0.0f;
        for (int d = 0; d < D_DIM; ++d) { float v = yn[d]; s += v * v; }
        yy = s;
    }
    __syncthreads();
    if (t < RDIM) {
        float u = 0.0f;
        for (int l = 0; l < RDIM; ++l)
            u += ws[WS_W + t * RDIM + l] * (hy[l] - ws[WS_T0 + l]);
        lm[t] = fmaxf(ws[WS_M + t] + u, 0.0f) + 1e-6f;
    }
    __syncthreads();
    if (t == 0) {
        float S = 0.0f;
        for (int j = 0; j < RDIM; ++j) S += lm[j];
        float ssq = 0.0f;
        for (int j = 0; j < RDIM; ++j) { lm[j] /= S; ssq += lm[j] * lm[j]; }
        float kn = (1.0f - ssq) / ws[WS_TRC] - 1.0f;
        float dsum = 0.0f, lgs = 0.0f;
        for (int j = 0; j < RDIM; ++j) {
            float dk = fmaxf(kn * lm[j], 0.0f) + REG_C;
            ws[WS_DIRI + (size_t)n * RDIM + j] = dk;
            ws[WS_HY   + (size_t)n * RDIM + j] = hy[j];
            dsum += dk;
            lgs  += lgammaf(dk);
        }
        ws[WS_YY + n] = yy;
        ws[WS_CQ + n] = lgammaf(dsum) - lgs;   // proposal log-normalizer
    }
}

// ---------------- Kernel C: sampling + softmax + WMMA moments ----------------
__global__ void __launch_bounds__(256)
LISA_main_kernel(const float* __restrict__ alpha,
                 const float* __restrict__ ws,
                 float* __restrict__ out) {
    const int n    = blockIdx.x;
    const int t    = threadIdx.x;       // 256 threads = 8 waves
    const int lane = t & 31;
    const int wave = t >> 5;

    // transposed, pre-converted f16 staging: row k holds sample index h
    __shared__ _Float16 zT16[RDIM * ZTS];    // z[h][k]   -> zT16[k*ZTS + h]
    __shared__ _Float16 wzT16[RDIM * ZTS];   // w[h]*z    -> wzT16[k*ZTS + h]
    __shared__ float wl[R_PAD];              // logits
    __shared__ float Gsh[RDIM * RDIM];
    __shared__ float dS[RDIM], aS[RDIM], bS[RDIM];
    __shared__ float red[256];
    __shared__ float zyacc[RDIM];
    __shared__ float zzred[32 * 8];
    __shared__ float cons[2];                // yy, (cp - cq[n])

    // ---- stage 0: load per-n constants ----
    __builtin_prefetch(&ws[WS_DIRI + (size_t)n * RDIM], 0, 0);
    __builtin_prefetch(&ws[WS_G], 0, 0);
    if (t < 2)   cons[t] = (t == 0) ? ws[WS_YY + n] : (ws[WS_CP] - ws[WS_CQ + n]);
    if (t < RDIM) {
        dS[t] = ws[WS_DIRI + (size_t)n * RDIM + t];
        aS[t] = alpha[t];
        bS[t] = ws[WS_HY + (size_t)n * RDIM + t];
        zyacc[t] = 0.0f;
    }
    for (int i = t; i < RDIM * RDIM; i += 256) Gsh[i] = ws[WS_G + i];
    if (t < 32 * 8) zzred[t] = 0.0f;
    __syncthreads();

    // ---- stage 1: generate samples (store f16 transposed) + logits ----
    const float yyv = cons[0], cpd = cons[1];
    for (int h = t; h < R_PAD; h += 256) {
        if (h < R_SAMP) {
            unsigned long long s =
                mix64(((unsigned long long)n << 32) ^ (unsigned long long)(h * 2654435761u) ^ 0xD1B54A32D192ED03ULL);
            float z[RDIM];
            float S = 0.0f;
            for (int kk = 0; kk < RDIM; ++kk) {
                z[kk] = gamma_sample(dS[kk], s);
                S += z[kk];
            }
            const float invS = 1.0f / S;
            for (int kk = 0; kk < RDIM; ++kk) {
                z[kk] = fmaxf(z[kk] * invS, 1e-12f);
                zT16[kk * ZTS + h] = (_Float16)z[kk];
            }
            // pyz via quadratic form: z^T G z - 2 b.z + yy
            float q = 0.0f, bz = 0.0f, sdl = 0.0f;
            for (int kk = 0; kk < RDIM; ++kk) {
                float gz = 0.0f;
                for (int ll = 0; ll < RDIM; ++ll) gz += Gsh[kk * RDIM + ll] * z[ll];
                q   += z[kk] * gz;
                bz  += bS[kk] * z[kk];
                sdl += (aS[kk] - dS[kk]) * logf(z[kk]);  // (pz - qz) data term
            }
            wl[h] = (q - 2.0f * bz + yyv) * (-0.5f / SIGMA_C) + sdl + cpd;
        } else {
            for (int kk = 0; kk < RDIM; ++kk) zT16[kk * ZTS + h] = (_Float16)0.0f;
            wl[h] = -1e30f;
        }
    }
    __syncthreads();

    // ---- stage 2: softmax over R ----
    float lmax = -3.0e38f;
    for (int h = t; h < R_PAD; h += 256) lmax = fmaxf(lmax, wl[h]);
    red[t] = lmax; __syncthreads();
    for (int s2 = 128; s2 > 0; s2 >>= 1) {
        if (t < s2) red[t] = fmaxf(red[t], red[t + s2]);
        __syncthreads();
    }
    const float mv = red[0]; __syncthreads();
    float lsum = 0.0f;
    for (int h = t; h < R_PAD; h += 256) lsum += expf(wl[h] - mv);
    red[t] = lsum; __syncthreads();
    for (int s2 = 128; s2 > 0; s2 >>= 1) {
        if (t < s2) red[t] += red[t + s2];
        __syncthreads();
    }
    const float winv = 1.0f / red[0]; __syncthreads();

    // ---- stage 3: weights -> wzT16 (f16, transposed) + zy accumulation ----
    {
        float acc[RDIM];
        for (int kk = 0; kk < RDIM; ++kk) acc[kk] = 0.0f;
        for (int h = t; h < R_PAD; h += 256) {
            const float w = expf(wl[h] - mv) * winv;
            for (int kk = 0; kk < RDIM; ++kk) {
                const float zv = (float)zT16[kk * ZTS + h];
                acc[kk] += w * zv;
                wzT16[kk * ZTS + h] = (_Float16)(w * zv);
            }
        }
        for (int kk = 0; kk < RDIM; ++kk) atomicAdd(&zyacc[kk], acc[kk]);
    }
    __syncthreads();
    if (t < RDIM) out[(size_t)n * RDIM + t] = zyacc[t];

    // ---- stage 4: zzy via WMMA: D += A(16x32) * B(32x16); 16 chunks, 2 per wave ----
    // A[M][K] = w[h0+K]*z[h0+K][M] = wzT16[M][h0+K] ; B[K][N] = zT16[N][h0+K]
    // f16 A layout: lanes 0-15 K in {0..7,16..23}; lanes 16-31 K in {8..15,24..31}
    // f16 B layout: lanes 0-15 K in {0..7,16..23}; lanes 16-31 K in {8..15,24..31} ; N = lane&15
    const int grp  = lane >> 4;
    const int Mrow = lane & 15;
    const _Float16* zrow  = &zT16[Mrow * ZTS];
    const _Float16* wzrow = &wzT16[Mrow * ZTS];

    v8f acc0 = {}, acc1 = {};
    {
        const int h0 = wave * 32;                 // chunk c = wave
        v8h alo = *(const v8h*)(wzrow + h0 + 8 * grp);
        v8h ahi = *(const v8h*)(wzrow + h0 + 16 + 8 * grp);
        v16h a  = __builtin_shufflevector(alo, ahi, 0,1,2,3,4,5,6,7,8,9,10,11,12,13,14,15);
        v8h blo = *(const v8h*)(zrow + h0 + 8 * grp);
        v8h bhi = *(const v8h*)(zrow + h0 + 16 + 8 * grp);
        v16h b  = __builtin_shufflevector(blo, bhi, 0,1,2,3,4,5,6,7,8,9,10,11,12,13,14,15);
        acc0 = __builtin_amdgcn_wmma_f32_16x16x32_f16(false, a, false, b,
                                                      (short)0, acc0, false, false);
    }
    {
        const int h0 = (wave + 8) * 32;           // chunk c = wave + 8
        v8h alo = *(const v8h*)(wzrow + h0 + 8 * grp);
        v8h ahi = *(const v8h*)(wzrow + h0 + 16 + 8 * grp);
        v16h a  = __builtin_shufflevector(alo, ahi, 0,1,2,3,4,5,6,7,8,9,10,11,12,13,14,15);
        v8h blo = *(const v8h*)(zrow + h0 + 8 * grp);
        v8h bhi = *(const v8h*)(zrow + h0 + 16 + 8 * grp);
        v16h b  = __builtin_shufflevector(blo, bhi, 0,1,2,3,4,5,6,7,8,9,10,11,12,13,14,15);
        acc1 = __builtin_amdgcn_wmma_f32_16x16x32_f16(false, a, false, b,
                                                      (short)0, acc1, false, false);
    }
    // cross-wave reduce of the 16x16 accumulator
    for (int v = 0; v < 8; ++v) atomicAdd(&zzred[lane * 8 + v], acc0[v] + acc1[v]);
    __syncthreads();
    if (wave == 0) {
        const size_t zz0 = (size_t)N_OBS * RDIM + (size_t)n * RDIM * RDIM;
        for (int v = 0; v < 8; ++v) {
            int M  = v + 8 * (lane >> 4);
            int Nc = lane & 15;
            out[zz0 + M * RDIM + Nc] = zzred[lane * 8 + v];
        }
    }
}

// ---------------- host launcher ----------------
extern "C" void kernel_launch(void* const* d_in, const int* in_sizes, int n_in,
                              void* d_out, int out_size, void* d_ws, size_t ws_size,
                              hipStream_t stream) {
    (void)in_sizes; (void)n_in; (void)out_size; (void)ws_size;
    const float* y     = (const float*)d_in[0];   // [N, D]
    const float* H     = (const float*)d_in[1];   // [D, RDIM]
    const float* alpha = (const float*)d_in[2];   // [RDIM]
    float* out = (float*)d_out;                   // zy [N,16] ++ zzy [N,16,16]
    float* ws  = (float*)d_ws;

    LISA_setup_kernel<<<1, 256, 0, stream>>>(H, alpha, ws);
    LISA_prep_kernel<<<N_OBS, 32, 0, stream>>>(y, H, ws);
    LISA_main_kernel<<<N_OBS, 256, 0, stream>>>(alpha, ws, out);
}
